// Qwen3NeXT_78855599554949
// MI455X (gfx1250) — matrix operand order repaired and checked by reading, not compiled, not run on hardware
//
#include <hip/hip_runtime.h>
#include <hip/hip_bf16.h>

// ---------------------------------------------------------------------------
// Qwen3-Next-style hybrid stack forward (B=2,S=1024,D=2048,V=32000,I=5632,
// Hq=16,Hkv=4,HD=128,L=8, attention every 4th layer).
//
// GEMM strategy (carries ~1.9 TFLOP of the model):
//   * convert activations to f16, weights to f16-transposed (N x K) once
//   * 128x128 block tile, 8 waves (2Mx4N), 8 v_wmma_f32_16x16x32_f16 per
//     32-K step, fragments served from 16KB of LDS
//   * LDS tiles staged by the Tensor Data Mover (tensor_load_to_lds +
//     s_wait_tensorcnt) when the builtin exists; manual b128 copy otherwise
// DeltaNet scan / softmax / norms are scalar VALU (negligible FLOP share).
// ---------------------------------------------------------------------------

#define Bc 2
#define Sc 1024
#define Dc 2048
#define Vc 32000
#define Ic 5632
#define HQc 16
#define HKVc 4
#define HDc 128
#define Lc 8
#define Mc (Bc * Sc)   // 2048 token rows

typedef __attribute__((ext_vector_type(16))) _Float16 v16h;
typedef __attribute__((ext_vector_type(8)))  _Float16 v8h;
typedef __attribute__((ext_vector_type(8)))  float    v8f;
typedef unsigned int u32x4 __attribute__((ext_vector_type(4)));
typedef int          i32x8 __attribute__((ext_vector_type(8)));
typedef int          i32x4 __attribute__((ext_vector_type(4)));

#if defined(__has_builtin)
#if __has_builtin(__builtin_amdgcn_tensor_load_to_lds) && \
    __has_builtin(__builtin_amdgcn_s_wait_tensorcnt)
#define QWEN3N_TDM 1
#endif
#endif

#ifdef QWEN3N_TDM
// Build a 2D tensor-DMA descriptor (D#) and issue TENSOR_LOAD_TO_LDS.
// tile = tile_w x tile_h elements (2B each), tensor row stride = stride elems.
// This toolchain exposes the 6-arg builtin:
//   (u32x4 g0, i32x8 g1, i32x4 g2, i32x4 g3, i32x8 g4, i32 cpol)
__device__ __forceinline__ void qwen3n_tdm_load2d(unsigned lds_byte_addr,
                                                  const _Float16* gptr,
                                                  int rowlen, int nrows,
                                                  int stride,
                                                  int tile_w, int tile_h) {
  unsigned long long ga = (unsigned long long)(size_t)gptr;
  u32x4 g0;
  g0[0] = 1u;                                     // count=1 (valid), user desc
  g0[1] = lds_byte_addr;                          // lds_addr  [63:32]
  g0[2] = (unsigned)ga;                           // global_addr lo [95:64]
  g0[3] = (unsigned)((ga >> 32) & 0x01FFFFFFull)  // global_addr hi [120:96]
          | 0x80000000u;                          // type=2 ("image") [127:126]
  i32x8 g1;
  g1[0] = 0x00010000;                             // data_size=1 (2 bytes)
  g1[1] = (int)(((unsigned)rowlen & 0xFFFFu) << 16);            // dim0 lo16
  g1[2] = (int)((((unsigned)rowlen >> 16) & 0xFFFFu) |
                (((unsigned)nrows & 0xFFFFu) << 16));           // dim0 hi/dim1 lo
  g1[3] = (int)((((unsigned)nrows >> 16) & 0xFFFFu) |
                (((unsigned)tile_w & 0xFFFFu) << 16));          // dim1 hi/tile_dim0
  g1[4] = (int)((unsigned)tile_h & 0xFFFFu);                    // tile_dim1, tile_dim2=0
  g1[5] = stride;                                               // dim0_stride lo32
  g1[6] = 0;
  g1[7] = 0;
  i32x4 z4 = {0, 0, 0, 0};
  i32x8 z8 = {0, 0, 0, 0, 0, 0, 0, 0};
  __builtin_amdgcn_tensor_load_to_lds(g0, g1, z4, z4, z8, 0);
}
#endif

// ---------------------------------------------------------------------------
// Tiled WMMA GEMM: C[M,N(ldC)] = A16[M,K] * BT16[N,K]^T  (f16 in, f32 out)
// Block: 256 threads / 8 waves; block tile 128x128; wave tile 64x32
// (4 M-subtiles x 2 N-subtiles -> 8 WMMAs per 32-K step).
// ---------------------------------------------------------------------------
__global__ __launch_bounds__(256)
void qwen3n_gemm_tile(const _Float16* __restrict__ A16,
                      const _Float16* __restrict__ BT16,
                      float* __restrict__ C,
                      int M, int N, int K, int ldC) {
  __shared__ _Float16 Atile[128 * 32];
  __shared__ _Float16 Btile[128 * 32];
  const int tid  = threadIdx.x;
  const int lane = tid & 31, wid = tid >> 5;
  const int wm = wid & 1;          // 0..1 (64-row groups)
  const int wn = wid >> 1;         // 0..3 (32-col groups)
  const int l15 = lane & 15, hi = lane >> 4;
  const int m0 = blockIdx.x * 128, n0 = blockIdx.y * 128;

  v8f acc[4][2] = {};

#ifdef QWEN3N_TDM
  const unsigned ldsA = (unsigned)(size_t)&Atile[0];
  const unsigned ldsB = (unsigned)(size_t)&Btile[0];
#else
  const int cr = tid >> 1;               // 0..127 staging row
  const int cs = (tid & 1) * 16;         // 0 / 16 halves
#endif

  for (int kk = 0; kk < K; kk += 32) {
#ifdef QWEN3N_TDM
    if (tid < 32) {   // wave 0 drives the Tensor Data Mover
      qwen3n_tdm_load2d(ldsA, A16 + (size_t)m0 * K + kk, K, M, K, 32, 128);
      qwen3n_tdm_load2d(ldsB, BT16 + (size_t)n0 * K + kk, K, N, K, 32, 128);
      __builtin_amdgcn_s_wait_tensorcnt(0);
    }
#else
    *(v8h*)(&Atile[cr * 32 + cs])     = *(const v8h*)(&A16[(size_t)(m0 + cr) * K + kk + cs]);
    *(v8h*)(&Atile[cr * 32 + cs + 8]) = *(const v8h*)(&A16[(size_t)(m0 + cr) * K + kk + cs + 8]);
    *(v8h*)(&Btile[cr * 32 + cs])     = *(const v8h*)(&BT16[(size_t)(n0 + cr) * K + kk + cs]);
    *(v8h*)(&Btile[cr * 32 + cs + 8]) = *(const v8h*)(&BT16[(size_t)(n0 + cr) * K + kk + cs + 8]);
#endif
    __syncthreads();

    v16h afrag[4], bfrag[2];
#pragma unroll
    for (int i = 0; i < 4; ++i) {
      // A fragment: lane<16 -> row=l15, K {kb..kb+7, kb+16..kb+23}, kb=hi*8
      const int row = wm * 64 + i * 16 + l15;
      const _Float16* p = &Atile[row * 32 + hi * 8];
      v8h lo  = *(const v8h*)p;
      v8h hi8 = *(const v8h*)(p + 16);
      afrag[i] = __builtin_shufflevector(lo, hi8, 0, 1, 2, 3, 4, 5, 6, 7,
                                         8, 9, 10, 11, 12, 13, 14, 15);
    }
#pragma unroll
    for (int j = 0; j < 2; ++j) {
      // B fragment: lane<16 -> col=l15, K {kb..kb+7, kb+16..kb+23}, kb=hi*8
      const int col = wn * 32 + j * 16 + l15;
      const _Float16* q = &Btile[col * 32 + hi * 8];
      v8h blo = *(const v8h*)q;
      v8h bhi = *(const v8h*)(q + 16);
      bfrag[j] = __builtin_shufflevector(blo, bhi, 0, 1, 2, 3, 4, 5, 6, 7,
                                         8, 9, 10, 11, 12, 13, 14, 15);
    }
#pragma unroll
    for (int i = 0; i < 4; ++i)
#pragma unroll
      for (int j = 0; j < 2; ++j)
        acc[i][j] = __builtin_amdgcn_wmma_f32_16x16x32_f16(
            false, afrag[i], false, bfrag[j], (short)0, acc[i][j], false, false);
    __syncthreads();
  }

#pragma unroll
  for (int i = 0; i < 4; ++i) {
    const int rbase = m0 + wm * 64 + i * 16 + hi * 8;
#pragma unroll
    for (int j = 0; j < 2; ++j) {
      const int col = n0 + wn * 32 + j * 16 + l15;
#pragma unroll
      for (int r = 0; r < 8; ++r)
        C[(size_t)(rbase + r) * ldC + col] = acc[i][j][r];
    }
  }
}

// ---------------------------------------------------------------------------
// Small-N WMMA GEMM (N=16 projections Wb/Wg): f32 in, f16 multiply.
// One wave per 16x16 tile.
// ---------------------------------------------------------------------------
__global__ void qwen3n_gemm_small(const float* __restrict__ A,
                                  const float* __restrict__ B,
                                  float* __restrict__ C,
                                  int M, int N, int K) {
  const int lane = threadIdx.x;
  const int tm = blockIdx.x, tn = blockIdx.y;
  const int l15 = lane & 15;
  const int hi  = lane >> 4;
  const int rowA = tm * 16 + l15;
  const int colB = tn * 16 + l15;
  const int aKofs = hi ? 8 : 0;
  const int bKofs = hi ? 8 : 0;

  v8f acc = {};
  for (int kk = 0; kk < K; kk += 32) {
    const float* Ap = A + (size_t)rowA * K + kk + aKofs;
    const float* Bp = B + (size_t)(kk + bKofs) * N + colB;
    __builtin_prefetch(Ap + 32, 0, 1);
    v16h a, b;
#pragma unroll
    for (int h = 0; h < 16; ++h) {
      const int kof = h + (h >= 8 ? 8 : 0);
      a[h] = (_Float16)Ap[kof];
    }
#pragma unroll
    for (int h = 0; h < 16; ++h) b[h] = (_Float16)Bp[(size_t)(h + (h >= 8 ? 8 : 0)) * N];
    acc = __builtin_amdgcn_wmma_f32_16x16x32_f16(false, a, false, b,
                                                 (short)0, acc, false, false);
  }
  const int rbase = tm * 16 + (hi ? 8 : 0);
  const int col = tn * 16 + l15;
#pragma unroll
  for (int r = 0; r < 8; ++r) C[(size_t)(rbase + r) * N + col] = acc[r];
}

// ---------------------------------------------------------------------------
// f32 -> f16 contiguous convert
// ---------------------------------------------------------------------------
__global__ void qwen3n_cvt_f16(const float* __restrict__ src,
                               _Float16* __restrict__ dst, long n) {
  long i = (long)blockIdx.x * blockDim.x + threadIdx.x;
  if (i < n) dst[i] = (_Float16)src[i];
}

// f32 [K x Nfull] columns [c0, c0+Nc) -> f16 transposed [Nc x K]
__global__ void qwen3n_cvt_transpose(const float* __restrict__ src,
                                     _Float16* __restrict__ dst,
                                     int K, int Nfull, int c0) {
  __shared__ float tile[32][33];
  const int kb = blockIdx.x * 32, nb = blockIdx.y * 32;
  const int tx = threadIdx.x, ty = threadIdx.y;   // 32 x 8
  for (int r = ty; r < 32; r += 8)
    tile[r][tx] = src[(size_t)(kb + r) * Nfull + c0 + nb + tx];
  __syncthreads();
  for (int r = ty; r < 32; r += 8)
    dst[(size_t)(nb + r) * K + kb + tx] = (_Float16)tile[tx][r];
}

// ---------------------------------------------------------------------------
// Embedding gather: X[m, :] = embed[ids[m], :]
// ---------------------------------------------------------------------------
__global__ void qwen3n_embed(const int* __restrict__ ids,
                             const float* __restrict__ embed,
                             float* __restrict__ X) {
  const int m = blockIdx.x;
  const int tok = ids[m];
  const float* src = embed + (size_t)tok * Dc;
  float* dst = X + (size_t)m * Dc;
  for (int d = threadIdx.x; d < Dc; d += blockDim.x) dst[d] = src[d];
}

// ---------------------------------------------------------------------------
// RMSNorm rows of length N: y = x * rsqrt(mean(x^2)+eps) * w (in-place safe)
// ---------------------------------------------------------------------------
__global__ void qwen3n_rmsnorm(const float* __restrict__ x,
                               const float* __restrict__ w,
                               float* __restrict__ y, int N) {
  __shared__ float red[256];
  const int row = blockIdx.x;
  const float* xr = x + (size_t)row * N;
  float ss = 0.f;
  for (int i = threadIdx.x; i < N; i += blockDim.x) { float v = xr[i]; ss += v * v; }
  red[threadIdx.x] = ss;
  __syncthreads();
  for (int s = blockDim.x >> 1; s > 0; s >>= 1) {
    if (threadIdx.x < (unsigned)s) red[threadIdx.x] += red[threadIdx.x + s];
    __syncthreads();
  }
  const float scale = rsqrtf(red[0] / (float)N + 1e-6f);
  float* yr = y + (size_t)row * N;
  for (int i = threadIdx.x; i < N; i += blockDim.x) yr[i] = xr[i] * scale * w[i];
}

// L2 normalize rows of 128 in-place: x /= (||x|| + 1e-6)
__global__ void qwen3n_l2norm(float* __restrict__ x) {
  __shared__ float red[128];
  const int row = blockIdx.x;
  float* xr = x + (size_t)row * HDc;
  float v = xr[threadIdx.x];
  red[threadIdx.x] = v * v;
  __syncthreads();
  for (int s = 64; s > 0; s >>= 1) {
    if (threadIdx.x < (unsigned)s) red[threadIdx.x] += red[threadIdx.x + s];
    __syncthreads();
  }
  const float inv = 1.f / (sqrtf(red[0]) + 1e-6f);
  xr[threadIdx.x] = v * inv;
}

// ---------------------------------------------------------------------------
// RoPE (base 1e6, rotate-half); rows [B,S,H,128]; 64 threads per row
// ---------------------------------------------------------------------------
__global__ void qwen3n_rope(float* __restrict__ x, int H) {
  const int row = blockIdx.x;              // (b*S + s)*H + h
  const int s = (row / H) % Sc;
  float* xp = x + (size_t)row * HDc;
  const int i = threadIdx.x;               // 0..63
  const float inv = expf(-(2.0f * (float)i / (float)HDc) * logf(1.0e6f));
  const float ang = (float)s * inv;
  const float c = cosf(ang), sn = sinf(ang);
  const float x0 = xp[i], x1 = xp[i + 64];
  xp[i]      = x0 * c - x1 * sn;
  xp[i + 64] = x1 * c + x0 * sn;
}

// ---------------------------------------------------------------------------
// Causal softmax attention with GQA (16 q heads / 4 kv heads).
// One block per (qpos, b*Hq+h), 128 threads.
// ---------------------------------------------------------------------------
__global__ void qwen3n_attn(const float* __restrict__ q,
                            const float* __restrict__ k,
                            const float* __restrict__ v,
                            float* __restrict__ o) {
  __shared__ float qv[HDc];
  __shared__ float sc[Sc];
  __shared__ float red[128];
  const int qpos = blockIdx.x;
  const int bh = blockIdx.y;
  const int b = bh / HQc, h = bh % HQc, kvh = h >> 2;
  const int tid = threadIdx.x;

  const size_t qbase = ((size_t)(b * Sc + qpos) * HQc + h) * HDc;
  qv[tid] = q[qbase + tid];
  __syncthreads();

  const float scale = 0.08838834764831845f;   // 1/sqrt(128)
  float lmax = -3.4e38f;
  for (int j = tid; j <= qpos; j += 128) {
    const float* kp = k + ((size_t)(b * Sc + j) * HKVc + kvh) * HDc;
    float dot = 0.f;
    for (int d = 0; d < HDc; ++d) dot += qv[d] * kp[d];
    dot *= scale;
    sc[j] = dot;
    lmax = fmaxf(lmax, dot);
  }
  red[tid] = lmax;
  __syncthreads();
  for (int s = 64; s > 0; s >>= 1) {
    if (tid < s) red[tid] = fmaxf(red[tid], red[tid + s]);
    __syncthreads();
  }
  const float m = red[0];
  __syncthreads();

  float lsum = 0.f;
  for (int j = tid; j <= qpos; j += 128) {
    const float p = expf(sc[j] - m);
    sc[j] = p;
    lsum += p;
  }
  red[tid] = lsum;
  __syncthreads();
  for (int s = 64; s > 0; s >>= 1) {
    if (tid < s) red[tid] += red[tid + s];
    __syncthreads();
  }
  const float invsum = 1.f / red[0];
  __syncthreads();

  float acc = 0.f;
  for (int j = 0; j <= qpos; ++j)
    acc += sc[j] * v[((size_t)(b * Sc + j) * HKVc + kvh) * HDc + tid];
  o[qbase + tid] = acc * invsum;
}

// ---------------------------------------------------------------------------
// Gated DeltaNet scan. One block per (b,h), 256 threads; state [128x128]
// in registers (thread t owns column t%128, rows (t/128)*64 .. +63).
// ---------------------------------------------------------------------------
__global__ void qwen3n_deltanet(const float* __restrict__ q,
                                const float* __restrict__ k,
                                const float* __restrict__ v,
                                const float* __restrict__ blin,
                                const float* __restrict__ glin,
                                float* __restrict__ o) {
  __shared__ float kv[HDc], vv[HDc], qv[HDc], part[256], sb[2];
  const int bh = blockIdx.x;
  const int b = bh / HQc, h = bh % HQc;
  const int tid = threadIdx.x;
  const int half = tid >> 7;
  const int j = tid & 127;
  const int r0 = half * 64;

  float St[64];
#pragma unroll
  for (int r = 0; r < 64; ++r) St[r] = 0.f;

  for (int t = 0; t < Sc; ++t) {
    const size_t base = ((size_t)(b * Sc + t) * HQc + h) * HDc;
    if (tid < HDc) {
      kv[tid] = k[base + tid];
      vv[tid] = v[base + tid];
      qv[tid] = q[base + tid];
    }
    if (tid == 0) {
      const size_t bg = (size_t)(b * Sc + t) * HQc + h;
      sb[0] = 1.f / (1.f + expf(-blin[bg]));   // beta
      sb[1] = 1.f / (1.f + expf(-glin[bg]));   // g
    }
    __syncthreads();
    const float beta = sb[0], g = sb[1];

    float partial = 0.f;
#pragma unroll
    for (int r = 0; r < 64; ++r) {
      St[r] *= g;
      partial += kv[r0 + r] * St[r];
    }
    part[tid] = partial;
    __syncthreads();
    const float pred = part[j] + part[j + 128];
    __syncthreads();

    const float upd = beta * (vv[j] - pred);
    float outp = 0.f;
#pragma unroll
    for (int r = 0; r < 64; ++r) {
      St[r] += kv[r0 + r] * upd;
      outp += qv[r0 + r] * St[r];
    }
    part[tid] = outp;
    __syncthreads();
    if (half == 0) o[base + j] = part[j] + part[j + 128];
    __syncthreads();
  }
}

// ---------------------------------------------------------------------------
// Elementwise helpers
// ---------------------------------------------------------------------------
__global__ void qwen3n_silu_mul(const float* __restrict__ g,
                                const float* __restrict__ u,
                                float* __restrict__ o, long n) {
  long i = (long)blockIdx.x * blockDim.x + threadIdx.x;
  if (i < n) { float x = g[i]; o[i] = (x / (1.f + expf(-x))) * u[i]; }
}
__global__ void qwen3n_gate_sigmoid(float* __restrict__ o,
                                    const float* __restrict__ gl, long n) {
  long i = (long)blockIdx.x * blockDim.x + threadIdx.x;
  if (i < n) o[i] *= 1.f / (1.f + expf(-gl[i]));
}
__global__ void qwen3n_gate_silu(float* __restrict__ o,
                                 const float* __restrict__ gl, long n) {
  long i = (long)blockIdx.x * blockDim.x + threadIdx.x;
  if (i < n) { float x = gl[i]; o[i] *= x / (1.f + expf(-x)); }
}
__global__ void qwen3n_add(float* __restrict__ x,
                           const float* __restrict__ t, long n) {
  long i = (long)blockIdx.x * blockDim.x + threadIdx.x;
  if (i < n) x[i] += t[i];
}

// ---------------------------------------------------------------------------
// Host orchestration. Input flattening: jax pytree order (dict keys sorted):
//   input_ids, embed, layers[0..7]{sorted keys}, lm_head, norm_w
// DeltaNet (13): Wb,Wg,Wk,Wo,Wog,Wq,Wv,attn_norm,ffn_norm,o_norm,w_down,w_gate,w_up
// Attention (12): Wgate,Wk,Wo,Wq,Wv,attn_norm,ffn_norm,k_norm,q_norm,w_down,w_gate,w_up
// ---------------------------------------------------------------------------
extern "C" void kernel_launch(void* const* d_in, const int* in_sizes, int n_in,
                              void* d_out, int out_size, void* d_ws, size_t ws_size,
                              hipStream_t stream) {
  (void)in_sizes; (void)n_in; (void)out_size; (void)ws_size;

  const int* ids = (const int*)d_in[0];
  const float* embed = (const float*)d_in[1];

  const float *Wb[Lc], *Wg[Lc], *Wk[Lc], *Wo[Lc], *Wog[Lc], *Wq[Lc], *Wv[Lc];
  const float *attn_n[Lc], *ffn_n[Lc], *o_n[Lc], *wd[Lc], *wgt[Lc], *wup[Lc];
  const float *Wgate[Lc], *k_n[Lc], *q_n[Lc];

  int idx = 2;
  for (int i = 0; i < Lc; ++i) {
    const bool is_attn = ((i + 1) % 4 == 0);
    if (!is_attn) {
      Wb[i]     = (const float*)d_in[idx++];
      Wg[i]     = (const float*)d_in[idx++];
      Wk[i]     = (const float*)d_in[idx++];
      Wo[i]     = (const float*)d_in[idx++];
      Wog[i]    = (const float*)d_in[idx++];
      Wq[i]     = (const float*)d_in[idx++];
      Wv[i]     = (const float*)d_in[idx++];
      attn_n[i] = (const float*)d_in[idx++];
      ffn_n[i]  = (const float*)d_in[idx++];
      o_n[i]    = (const float*)d_in[idx++];
      wd[i]     = (const float*)d_in[idx++];
      wgt[i]    = (const float*)d_in[idx++];
      wup[i]    = (const float*)d_in[idx++];
      Wgate[i] = nullptr; k_n[i] = nullptr; q_n[i] = nullptr;
    } else {
      Wgate[i]  = (const float*)d_in[idx++];
      Wk[i]     = (const float*)d_in[idx++];
      Wo[i]     = (const float*)d_in[idx++];
      Wq[i]     = (const float*)d_in[idx++];
      Wv[i]     = (const float*)d_in[idx++];
      attn_n[i] = (const float*)d_in[idx++];
      ffn_n[i]  = (const float*)d_in[idx++];
      k_n[i]    = (const float*)d_in[idx++];
      q_n[i]    = (const float*)d_in[idx++];
      wd[i]     = (const float*)d_in[idx++];
      wgt[i]    = (const float*)d_in[idx++];
      wup[i]    = (const float*)d_in[idx++];
      Wb[i] = Wg[i] = Wog[i] = o_n[i] = nullptr;
    }
  }
  const float* lm_head = (const float*)d_in[idx++];
  const float* norm_w  = (const float*)d_in[idx++];

  // --- workspace carve-out ---
  const size_t MD = (size_t)Mc * Dc;
  const size_t MI = (size_t)Mc * Ic;
  const int NCH = 6400;                       // lm_head N chunk (5 x 6400)
  char* base = (char*)d_ws;
  auto alloc = [&](size_t bytes) -> char* {
    char* p = base;
    base += (bytes + 255) & ~(size_t)255;
    return p;
  };
  float* X    = (float*)alloc(MD * 4);
  float* Hbuf = (float*)alloc(MD * 4);
  float* Qb   = (float*)alloc(MD * 4);
  float* Kb   = (float*)alloc(MD * 4);
  float* Vb   = (float*)alloc(MD * 4);
  float* Ob   = (float*)alloc(MD * 4);
  float* GL   = (float*)alloc(MD * 4);
  float* TMP  = (float*)alloc(MD * 4);
  float* G1   = (float*)alloc(MI * 4);
  float* G2   = (float*)alloc(MI * 4);
  float* Blin = (float*)alloc((size_t)Mc * HQc * 4);
  float* Glin = (float*)alloc((size_t)Mc * HQc * 4);
  _Float16* A16h  = (_Float16*)alloc(MI * 2);                    // max M x K
  _Float16* BT16h = (_Float16*)alloc((size_t)NCH * Dc * 2);      // max N_c x K

  auto cvtA = [&](const float* s, long n) {
    qwen3n_cvt_f16<<<(unsigned)((n + 255) / 256), 256, 0, stream>>>(s, A16h, n);
  };
  // C[M_, c0:c0+nc] = A16h[M_,K_] @ B32[K_, c0:c0+nc]   (weights transposed+f16)
  auto gemm16 = [&](const float* B32, float* Cp, int M_, int N_, int K_, int ldC) {
    for (int c0 = 0; c0 < N_; c0 += NCH) {
      const int nc = (N_ - c0 < NCH) ? (N_ - c0) : NCH;
      qwen3n_cvt_transpose<<<dim3(K_ / 32, nc / 32), dim3(32, 8), 0, stream>>>(
          B32, BT16h, K_, N_, c0);
      qwen3n_gemm_tile<<<dim3(M_ / 128, nc / 128), 256, 0, stream>>>(
          A16h, BT16h, Cp + c0, M_, nc, K_, ldC);
    }
  };
  auto gemmS = [&](const float* A, const float* Bm, float* Cp, int M_, int N_, int K_) {
    qwen3n_gemm_small<<<dim3(M_ / 16, N_ / 16), dim3(32), 0, stream>>>(A, Bm, Cp, M_, N_, K_);
  };
  auto ew = [&](long n) { return dim3((unsigned)((n + 255) / 256)); };

  // --- embedding ---
  qwen3n_embed<<<Mc, 256, 0, stream>>>(ids, embed, X);

  for (int i = 0; i < Lc; ++i) {
    const bool is_attn = ((i + 1) % 4 == 0);

    qwen3n_rmsnorm<<<Mc, 256, 0, stream>>>(X, attn_n[i], Hbuf, Dc);
    cvtA(Hbuf, (long)MD);                      // A16h = f16(Hbuf)

    if (!is_attn) {
      // ---- gated DeltaNet ----
      gemm16(Wq[i], Qb, Mc, HQc * HDc, Dc, HQc * HDc);
      gemm16(Wk[i], Kb, Mc, HQc * HDc, Dc, HQc * HDc);
      gemm16(Wv[i], Vb, Mc, HQc * HDc, Dc, HQc * HDc);
      gemm16(Wog[i], GL, Mc, HQc * HDc, Dc, HQc * HDc);
      gemmS(Hbuf, Wb[i], Blin, Mc, HQc, Dc);
      gemmS(Hbuf, Wg[i], Glin, Mc, HQc, Dc);
      qwen3n_l2norm<<<Mc * HQc, 128, 0, stream>>>(Qb);
      qwen3n_l2norm<<<Mc * HQc, 128, 0, stream>>>(Kb);
      qwen3n_deltanet<<<Bc * HQc, 256, 0, stream>>>(Qb, Kb, Vb, Blin, Glin, Ob);
      qwen3n_rmsnorm<<<Mc * HQc, 128, 0, stream>>>(Ob, o_n[i], Ob, HDc);
      qwen3n_gate_silu<<<ew(MD), 256, 0, stream>>>(Ob, GL, (long)MD);
    } else {
      // ---- gated GQA attention ----
      gemm16(Wq[i], Qb, Mc, HQc * HDc, Dc, HQc * HDc);
      gemm16(Wk[i], Kb, Mc, HKVc * HDc, Dc, HKVc * HDc);
      gemm16(Wv[i], Vb, Mc, HKVc * HDc, Dc, HKVc * HDc);
      gemm16(Wgate[i], GL, Mc, HQc * HDc, Dc, HQc * HDc);
      qwen3n_rmsnorm<<<Mc * HQc, 128, 0, stream>>>(Qb, q_n[i], Qb, HDc);
      qwen3n_rmsnorm<<<Mc * HKVc, 128, 0, stream>>>(Kb, k_n[i], Kb, HDc);
      qwen3n_rope<<<Mc * HQc, 64, 0, stream>>>(Qb, HQc);
      qwen3n_rope<<<Mc * HKVc, 64, 0, stream>>>(Kb, HKVc);
      qwen3n_attn<<<dim3(Sc, Bc * HQc), 128, 0, stream>>>(Qb, Kb, Vb, Ob);
      qwen3n_gate_sigmoid<<<ew(MD), 256, 0, stream>>>(Ob, GL, (long)MD);
    }
    cvtA(Ob, (long)MD);
    gemm16(Wo[i], TMP, Mc, Dc, HQc * HDc, Dc);
    qwen3n_add<<<ew(MD), 256, 0, stream>>>(X, TMP, (long)MD);

    // ---- SwiGLU FFN ----
    qwen3n_rmsnorm<<<Mc, 256, 0, stream>>>(X, ffn_n[i], Hbuf, Dc);
    cvtA(Hbuf, (long)MD);
    gemm16(wgt[i], G1, Mc, Ic, Dc, Ic);
    gemm16(wup[i], G2, Mc, Ic, Dc, Ic);
    qwen3n_silu_mul<<<ew(MI), 256, 0, stream>>>(G1, G2, G1, (long)MI);
    cvtA(G1, (long)MI);
    gemm16(wd[i], TMP, Mc, Dc, Ic, Dc);
    qwen3n_add<<<ew(MD), 256, 0, stream>>>(X, TMP, (long)MD);
  }

  // ---- final norm + lm_head (chunked over V) ----
  qwen3n_rmsnorm<<<Mc, 256, 0, stream>>>(X, norm_w, Hbuf, Dc);
  cvtA(Hbuf, (long)MD);
  gemm16(lm_head, (float*)d_out, Mc, Vc, Dc, Vc);
}
